// TripletMarginLossWithMining_11355893531292
// MI455X (gfx1250) — matrix operand order repaired and checked by reading, not compiled, not run on hardware
//
#include <hip/hip_runtime.h>

typedef _Float16 f16;
typedef __attribute__((ext_vector_type(4)))  _Float16 v4h;
typedef __attribute__((ext_vector_type(8)))  _Float16 v8h;
typedef __attribute__((ext_vector_type(16))) _Float16 v16h;
typedef __attribute__((ext_vector_type(8)))  float    v8f;
typedef int v4i_ __attribute__((vector_size(16)));

#define D_DIM 512
#define BS_STRIDE 520   // 512 + 8 f16 pad -> row stride == 4 DWORDs mod 64 banks

#if __has_builtin(__builtin_amdgcn_global_load_async_to_lds_b128)
#define HAS_ASYNC 1
#else
#define HAS_ASYNC 0
#endif

template <int N>
static __device__ __forceinline__ void waitAsync() {
#if HAS_ASYNC
#if __has_builtin(__builtin_amdgcn_s_wait_asynccnt)
  __builtin_amdgcn_s_wait_asynccnt(N);
#else
  asm volatile("s_wait_asynccnt %0" ::"n"(N) : "memory");
#endif
#endif
}

#if HAS_ASYNC
static __device__ __forceinline__ void asyncCopy16(const f16* gsrc, f16* ldst) {
  __builtin_amdgcn_global_load_async_to_lds_b128(
      (__attribute__((address_space(1))) v4i_*)(void*)gsrc,
      (__attribute__((address_space(3))) v4i_*)(void*)ldst,
      0, 0);
}
#endif

static __device__ __forceinline__ float waveReduceSum(float x) {
#pragma unroll
  for (int off = 16; off > 0; off >>= 1) x += __shfl_xor(x, off, 32);
  return x;
}

// ---------------------------------------------------------------------------
// Normalize anchors (fp32 -> f16) and compute pos_dist in full fp32.
// One wave per row; lane handles 16 elements as 4x float4.
// ---------------------------------------------------------------------------
__global__ __launch_bounds__(256) void norm_anchor_kernel(
    const float* __restrict__ anchors, const float* __restrict__ positives,
    f16* __restrict__ aN, float* __restrict__ posd, int rows)
{
  const int wave = threadIdx.x >> 5;
  const int lane = threadIdx.x & 31;
  const int row  = blockIdx.x * 8 + wave;
  if (row >= rows) return;

  const float4* ap = (const float4*)(anchors   + (size_t)row * D_DIM);
  const float4* pp = (const float4*)(positives + (size_t)row * D_DIM);
  float4 av[4], pv[4];
  float sa = 0.f, sp = 0.f, sap = 0.f;
#pragma unroll
  for (int c = 0; c < 4; ++c) {
    av[c] = ap[lane + c * 32];
    pv[c] = pp[lane + c * 32];
    sa  += av[c].x*av[c].x + av[c].y*av[c].y + av[c].z*av[c].z + av[c].w*av[c].w;
    sp  += pv[c].x*pv[c].x + pv[c].y*pv[c].y + pv[c].z*pv[c].z + pv[c].w*pv[c].w;
    sap += av[c].x*pv[c].x + av[c].y*pv[c].y + av[c].z*pv[c].z + av[c].w*pv[c].w;
  }
  sa = waveReduceSum(sa); sp = waveReduceSum(sp); sap = waveReduceSum(sap);
  const float na  = fmaxf(sqrtf(sa), 1e-12f);
  const float np  = fmaxf(sqrtf(sp), 1e-12f);
  const float inv = 1.f / na;

  v4h* dst = (v4h*)(aN + (size_t)row * D_DIM);
#pragma unroll
  for (int c = 0; c < 4; ++c) {
    v4h o;
    o[0] = (f16)(av[c].x * inv); o[1] = (f16)(av[c].y * inv);
    o[2] = (f16)(av[c].z * inv); o[3] = (f16)(av[c].w * inv);
    dst[lane + c * 32] = o;
  }
  if (lane == 0) posd[row] = 1.f - sap / (na * np);
}

// ---------------------------------------------------------------------------
// Normalize embedding rows (fp32 -> f16). One wave per row.
// ---------------------------------------------------------------------------
__global__ __launch_bounds__(256) void norm_embed_kernel(
    const float* __restrict__ src, f16* __restrict__ dst, int rows)
{
  const int wave = threadIdx.x >> 5;
  const int lane = threadIdx.x & 31;
  const int row  = blockIdx.x * 8 + wave;
  if (row >= rows) return;

  const float4* sp = (const float4*)(src + (size_t)row * D_DIM);
  float4 xv[4];
  float ss = 0.f;
#pragma unroll
  for (int c = 0; c < 4; ++c) {
    xv[c] = sp[lane + c * 32];
    ss += xv[c].x*xv[c].x + xv[c].y*xv[c].y + xv[c].z*xv[c].z + xv[c].w*xv[c].w;
  }
  ss = waveReduceSum(ss);
  const float inv = 1.f / fmaxf(sqrtf(ss), 1e-12f);

  v4h* dp = (v4h*)(dst + (size_t)row * D_DIM);
#pragma unroll
  for (int c = 0; c < 4; ++c) {
    v4h o;
    o[0] = (f16)(xv[c].x * inv); o[1] = (f16)(xv[c].y * inv);
    o[2] = (f16)(xv[c].z * inv); o[3] = (f16)(xv[c].w * inv);
    dp[lane + c * 32] = o;
  }
}

// ---------------------------------------------------------------------------
// Fused GEMM + masked row-max.
// Block: 256 threads = 8 waves. Block tile: M=128 rows x N=512 cols, K=512.
// Each wave: 16 rows; full A panel (16x512 f16) held in 128 VGPRs.
// B staged 32 cols at a time via async global->LDS copies, double buffered
// (ASYNCcnt + s_wait_asynccnt + barriers) so the copy of stage s+1 overlaps
// the WMMA chain of stage s. Rows past N are skipped; stale LDS garbage only
// reaches columns that the epilogue masks anyway.
// partial[m * nChunks + chunk] = max_{n in chunk, n != label[m], n < N} a_m.e_n
// ---------------------------------------------------------------------------
__global__ __launch_bounds__(256) void gemm_max_kernel(
    const f16* __restrict__ aN, const f16* __restrict__ eN,
    const int* __restrict__ labels, float* __restrict__ partial,
    int Nn, int nChunks)
{
#if HAS_ASYNC
  __shared__ f16 Bs[2][32 * BS_STRIDE];   // ~66.5 KB (double buffered)
#else
  __shared__ f16 Bs[1][32 * BS_STRIDE];   // ~33 KB
#endif

  const int tid  = threadIdx.x;
  const int wave = tid >> 5;
  const int lane = tid & 31;
  const int nl   = lane & 15;
  const int half = lane >> 4;
  const int chunk      = blockIdx.x;
  const int mbase      = blockIdx.y * 128 + wave * 16;
  const int chunkStart = chunk * 512;

  // ---- A fragments: contiguous K order (k = 16*half + elem), same as B ----
  // lanes 0-15 (half=0): elems 0..7 = K t*32+0..7,   elems 8..15 = K t*32+8..15
  // lanes 16-31(half=1): elems 0..7 = K t*32+16..23, elems 8..15 = K t*32+24..31
  v16h areg[16];
  {
    const f16* arow = aN + (size_t)(mbase + nl) * D_DIM;
#pragma unroll
    for (int t = 0; t < 16; ++t) {
      v8h lo = *(const v8h*)(arow + t * 32 + half * 16);
      v8h hi = *(const v8h*)(arow + t * 32 + half * 16 + 8);
#pragma unroll
      for (int j = 0; j < 8; ++j) { areg[t][j] = lo[j]; areg[t][j + 8] = hi[j]; }
    }
  }

  // labels for the 8 rows this lane's C fragment covers (M = mbase + half*8 + v)
  int labv[8];
#pragma unroll
  for (int v = 0; v < 8; ++v) labv[v] = labels[mbase + half * 8 + v];

  float rmax[8];
#pragma unroll
  for (int v = 0; v < 8; ++v) rmax[v] = -__builtin_inff();

#if HAS_ASYNC
  // issue stage s (32 rows x 512 f16, 2048 x 16B chunks, 8 per thread)
  auto issueStage = [&](int s) {
    const int colBase = chunkStart + s * 32;
    f16* base = &Bs[s & 1][0];
#pragma unroll
    for (int i = 0; i < 8; ++i) {
      const int idx = tid + i * 256;
      const int r   = idx >> 6;
      const int c   = idx & 63;
      const int gn  = colBase + r;
      if (gn < Nn)
        asyncCopy16(eN + (size_t)gn * D_DIM + c * 8, base + r * BS_STRIDE + c * 8);
    }
  };
  issueStage(0);
#endif

  for (int s = 0; s < 16; ++s) {           // 16 stages x 32 cols = 512-col chunk
    const int colBase = chunkStart + s * 32;
#if HAS_ASYNC
    if (s + 1 < 16) { issueStage(s + 1); waitAsync<8>(); }
    else            { waitAsync<0>(); }
    __syncthreads();                       // all waves' stage-s data visible
    const f16* bbuf = &Bs[s & 1][0];
#else
    __syncthreads();
#pragma unroll
    for (int i = 0; i < 8; ++i) {
      const int idx = tid + i * 256;
      const int r   = idx >> 6;
      const int c   = idx & 63;
      const int gn  = colBase + r;
      uint4 val = {0u, 0u, 0u, 0u};
      if (gn < Nn) val = *(const uint4*)(eN + (size_t)gn * D_DIM + c * 8);
      *(uint4*)(&Bs[0][r * BS_STRIDE + c * 8]) = val;
    }
    __syncthreads();
    const f16* bbuf = &Bs[0][0];
#endif

    v8f acc0 = {0.f,0.f,0.f,0.f,0.f,0.f,0.f,0.f};
    v8f acc1 = {0.f,0.f,0.f,0.f,0.f,0.f,0.f,0.f};
    // B 32x16 f16 layout: lane nl = column, half selects K 0-15 / 16-31
    const f16* b0 = bbuf + (nl)      * BS_STRIDE + half * 16;
    const f16* b1 = bbuf + (16 + nl) * BS_STRIDE + half * 16;
#pragma unroll
    for (int t = 0; t < 16; ++t) {
      v8h q0 = *(const v8h*)(b0 + t * 32);
      v8h q1 = *(const v8h*)(b0 + t * 32 + 8);
      v8h q2 = *(const v8h*)(b1 + t * 32);
      v8h q3 = *(const v8h*)(b1 + t * 32 + 8);
      v16h bf0, bf1;
#pragma unroll
      for (int j = 0; j < 8; ++j) {
        bf0[j] = q0[j]; bf0[j + 8] = q1[j];
        bf1[j] = q2[j]; bf1[j + 8] = q3[j];
      }
      acc0 = __builtin_amdgcn_wmma_f32_16x16x32_f16(false, areg[t], false, bf0,
                                                    (short)0, acc0, false, false);
      acc1 = __builtin_amdgcn_wmma_f32_16x16x32_f16(false, areg[t], false, bf1,
                                                    (short)0, acc1, false, false);
    }

    // C layout: n = lane&15 within subtile, m = mbase + half*8 + v
    const int n0 = colBase + nl;
    const int n1 = colBase + 16 + nl;
#pragma unroll
    for (int v = 0; v < 8; ++v) {
      const float d0 = (n0 < Nn && n0 != labv[v]) ? acc0[v] : -__builtin_inff();
      const float d1 = (n1 < Nn && n1 != labv[v]) ? acc1[v] : -__builtin_inff();
      rmax[v] = fmaxf(rmax[v], fmaxf(d0, d1));
    }
    __syncthreads();                       // buffer free before next overwrite
  }

  // max-reduce across the 16 lanes of each half, lane nl==0 writes
#pragma unroll
  for (int v = 0; v < 8; ++v) {
    float x = rmax[v];
#pragma unroll
    for (int off = 8; off > 0; off >>= 1) x = fmaxf(x, __shfl_xor(x, off, 32));
    if (nl == 0)
      partial[(size_t)(mbase + half * 8 + v) * nChunks + chunk] = x;
  }
}

// ---------------------------------------------------------------------------
// Per-row loss + deterministic two-stage mean.
// ---------------------------------------------------------------------------
__global__ __launch_bounds__(256) void rowloss_kernel(
    const float* __restrict__ partial, const float* __restrict__ posd,
    float* __restrict__ blockSums, int nChunks)
{
  __shared__ float red[256];
  const int m = blockIdx.x * 256 + threadIdx.x;
  float mx = -__builtin_inff();
  const float* pr = partial + (size_t)m * nChunks;
  for (int c = 0; c < nChunks; ++c) mx = fmaxf(mx, pr[c]);
  const float negd = 1.f - mx;
  const float loss = fmaxf(posd[m] - negd + 0.5f, 0.f);
  red[threadIdx.x] = loss;
  __syncthreads();
  for (int s = 128; s > 0; s >>= 1) {
    if (threadIdx.x < s) red[threadIdx.x] += red[threadIdx.x + s];
    __syncthreads();
  }
  if (threadIdx.x == 0) blockSums[blockIdx.x] = red[0];
}

__global__ void final_reduce_kernel(const float* __restrict__ blockSums,
                                    float* __restrict__ out, int n, float invB)
{
  if (threadIdx.x == 0 && blockIdx.x == 0) {
    float s = 0.f;
    for (int i = 0; i < n; ++i) s += blockSums[i];
    out[0] = s * invB;
  }
}

extern "C" void kernel_launch(void* const* d_in, const int* in_sizes, int n_in,
                              void* d_out, int out_size, void* d_ws, size_t ws_size,
                              hipStream_t stream)
{
  (void)n_in; (void)out_size; (void)ws_size;
  const float* anchors   = (const float*)d_in[0];
  const float* positives = (const float*)d_in[1];
  const float* allEmb    = (const float*)d_in[2];
  const int*   labels    = (const int*)d_in[3];

  const int B = in_sizes[3];              // 4096
  const int N = in_sizes[2] / D_DIM;      // 50000
  const int nChunks    = (N + 511) / 512; // 98
  const int mBlocks    = B / 128;         // 32
  const int lossBlocks = B / 256;         // 16

  char* w = (char*)d_ws;
  size_t off = 0;
  auto alloc = [&](size_t bytes) -> void* {
    void* p = w + off;
    off = (off + bytes + 255) & ~(size_t)255;
    return p;
  };
  f16*   aN      = (f16*)  alloc((size_t)B * D_DIM * sizeof(f16));
  f16*   eN      = (f16*)  alloc((size_t)N * D_DIM * sizeof(f16));
  float* posd    = (float*)alloc((size_t)B * sizeof(float));
  float* partial = (float*)alloc((size_t)B * nChunks * sizeof(float));
  float* bsum    = (float*)alloc((size_t)lossBlocks * sizeof(float));

  norm_anchor_kernel<<<dim3((B + 7) / 8), dim3(256), 0, stream>>>(
      anchors, positives, aN, posd, B);
  norm_embed_kernel<<<dim3((N + 7) / 8), dim3(256), 0, stream>>>(
      allEmb, eN, N);
  gemm_max_kernel<<<dim3(nChunks, mBlocks), dim3(256), 0, stream>>>(
      aN, eN, labels, partial, N, nChunks);
  rowloss_kernel<<<dim3(lossBlocks), dim3(256), 0, stream>>>(
      partial, posd, bsum, nChunks);
  final_reduce_kernel<<<dim3(1), dim3(32), 0, stream>>>(
      bsum, (float*)d_out, lossBlocks, 1.0f / (float)B);
}
